// GRULayer_24154896073420
// MI455X (gfx1250) — compile-verified, matrix operand order checked by reading, not run on hardware
//
#include <hip/hip_runtime.h>
#include <hip/hip_bf16.h>

// ---------------------------------------------------------------------------
// GRU layer for MI455X (gfx1250, wave32, WMMA).
//   B=64, S=512, D=512, H=1024, 3H=3072.
// Phase 1: convert weights/x to bf16 (weights transposed, K contiguous).
// Phase 2: x_proj = x @ W_in + b_in as one big bf16 WMMA GEMM (fp32 accum).
// Phase 3: ONE persistent kernel runs all 512 recurrent steps:
//            - 64 waves (32 blocks x 2 waves), wave = one 16-col tile of H,
//              computing all three gate panels (12 WMMA accumulators).
//            - W_h panels staged once into LDS (ds_load_b128 fragments).
//            - fp32 hidden state held in registers per wave; bf16 copy of h
//              ping-pongs through workspace for the next step's A operand.
//            - device-wide generation barrier between steps (agent atomics).
// ---------------------------------------------------------------------------

typedef __bf16 v16bf __attribute__((ext_vector_type(16)));
typedef float  v8f   __attribute__((ext_vector_type(8)));

#define GRU_B   64
#define GRU_S   512
#define GRU_D   512
#define GRU_H   1024
#define GRU_3H  3072

#define SCAN_BLOCKS 32
#define LDSK        1032                 // padded row stride (bank-conflict-free)
#define PANEL_ROWS  48                   // 3 gates * 16 columns per wave
#define SCAN_LDS_BYTES (2 * PANEL_ROWS * LDSK * 2)   // 198,144 B < 320 KB

union Frag16 { v16bf v; uint4 q[2]; };
union Acc8   { v8f v; float f[8]; };

// Load one 16x32 bf16 WMMA operand tile from a row-major matrix (K contig).
// ISA lane layout (16-bit A/B, 16x32): row/col = lane%16;
//   lanes 0-15 hold K {k0..k0+7, k0+16..k0+23}, lanes 16-31 hold K {+8,+24}.
__device__ __forceinline__ v16bf load_tile16(const __bf16* __restrict__ base,
                                             int ld, int row0, int k0, int lane) {
    int r = row0 + (lane & 15);
    int k = k0 + ((lane >> 4) << 3);
    const __bf16* p = base + (size_t)r * ld + k;
    Frag16 f;
    f.q[0] = *(const uint4*)(p);        // K = k .. k+7
    f.q[1] = *(const uint4*)(p + 16);   // K = k+16 .. k+23
    return f.v;
}

// Same fragment gather from an LDS-resident panel (row stride LDSK).
__device__ __forceinline__ v16bf lds_tile16(const __bf16* lds_base,
                                            int row0, int k0, int lane) {
    const __bf16* p = lds_base + (size_t)(row0 + (lane & 15)) * LDSK
                               + k0 + ((lane >> 4) << 3);
    Frag16 f;
    f.q[0] = *(const uint4*)(p);
    f.q[1] = *(const uint4*)(p + 16);
    return f.v;
}

__device__ __forceinline__ v8f wmma_bf16(v16bf a, v16bf b, v8f c) {
    return __builtin_amdgcn_wmma_f32_16x16x32_bf16(
        /*neg_a=*/false, a, /*neg_b=*/false, b,
        /*c_mod=*/(short)0, c, /*reuse_a=*/false, /*reuse_b=*/false);
}

// ---------------------------------------------------------------------------
// Conversion kernels
// ---------------------------------------------------------------------------
__global__ void convert_bf16_kernel(const float* __restrict__ in,
                                    __bf16* __restrict__ out, size_t n) {
    size_t i = (size_t)blockIdx.x * blockDim.x + threadIdx.x;
    if (i < n) out[i] = (__bf16)in[i];
}

// out[n*K + k] = bf16(in[k*N + n])  : (K x N) fp32 -> (N x K) bf16 (K contig)
__global__ void transpose_bf16_kernel(const float* __restrict__ in,
                                      __bf16* __restrict__ out, int K, int N) {
    size_t i = (size_t)blockIdx.x * blockDim.x + threadIdx.x;
    if (i < (size_t)K * N) {
        int n = (int)(i / K);
        int k = (int)(i % K);
        out[i] = (__bf16)in[(size_t)k * N + n];
    }
}

// ---------------------------------------------------------------------------
// Phase 2: x_proj[s*B + b][n] = sum_d xb[b*S+s][d] * W_in[d][n] + b_in[n]
// Block = 128 threads (4 waves), wave = 32x32 C tile. Grid (512, 48).
// ---------------------------------------------------------------------------
__global__ __launch_bounds__(128) void
xproj_gemm_kernel(const __bf16* __restrict__ xb,     // [B*S][D] bf16
                  const __bf16* __restrict__ wint,   // [3H][D]  bf16
                  const float*  __restrict__ b_in,   // [3H]
                  float*        __restrict__ xproj)  // [S][B][3H]
{
    const int lane = threadIdx.x & 31;
    const int w    = threadIdx.x >> 5;
    const int m0   = blockIdx.x * 64 + (w >> 1) * 32;
    const int n0   = blockIdx.y * 64 + (w & 1) * 32;

    v8f acc[2][2] = {};

    for (int k0 = 0; k0 < GRU_D; k0 += 32) {
        v16bf a0 = load_tile16(xb,   GRU_D, m0,      k0, lane);
        v16bf a1 = load_tile16(xb,   GRU_D, m0 + 16, k0, lane);
        v16bf b0 = load_tile16(wint, GRU_D, n0,      k0, lane);
        v16bf b1 = load_tile16(wint, GRU_D, n0 + 16, k0, lane);
        acc[0][0] = wmma_bf16(a0, b0, acc[0][0]);
        acc[0][1] = wmma_bf16(a0, b1, acc[0][1]);
        acc[1][0] = wmma_bf16(a1, b0, acc[1][0]);
        acc[1][1] = wmma_bf16(a1, b1, acc[1][1]);
    }

    #pragma unroll
    for (int mi = 0; mi < 2; ++mi) {
        #pragma unroll
        for (int ni = 0; ni < 2; ++ni) {
            Acc8 c; c.v = acc[mi][ni];
            int n = n0 + ni * 16 + (lane & 15);
            float bias = b_in[n];
            #pragma unroll
            for (int v = 0; v < 8; ++v) {
                int m = m0 + mi * 16 + v + ((lane >> 4) << 3);
                int b = m >> 9;          // m / S
                int s = m & (GRU_S - 1); // m % S
                xproj[((size_t)s * GRU_B + b) * GRU_3H + n] = c.f[v] + bias;
            }
        }
    }
}

// ---------------------------------------------------------------------------
// Device-wide generation barrier for the persistent scan kernel.
// ---------------------------------------------------------------------------
__device__ __forceinline__ void grid_barrier(unsigned* bar_count, unsigned* bar_gen,
                                             unsigned nblocks, unsigned step) {
    __syncthreads();
    if (threadIdx.x == 0) {
        __threadfence();  // release this block's stores device-wide
        unsigned arrived = __hip_atomic_fetch_add(bar_count, 1u, __ATOMIC_ACQ_REL,
                                                  __HIP_MEMORY_SCOPE_AGENT);
        if (arrived == nblocks - 1) {
            __hip_atomic_store(bar_count, 0u, __ATOMIC_RELAXED,
                               __HIP_MEMORY_SCOPE_AGENT);
            __hip_atomic_store(bar_gen, step + 1u, __ATOMIC_RELEASE,
                               __HIP_MEMORY_SCOPE_AGENT);
        } else {
            while (__hip_atomic_load(bar_gen, __ATOMIC_ACQUIRE,
                                     __HIP_MEMORY_SCOPE_AGENT) <= step)
                __builtin_amdgcn_s_sleep(2);
        }
    }
    __syncthreads();
    __threadfence();  // acquire side: discard stale cached peer data
}

// ---------------------------------------------------------------------------
// Phase 3: persistent scan. 32 blocks x 64 threads; wave = column tile j.
// ---------------------------------------------------------------------------
__global__ __launch_bounds__(64) void
gru_scan_kernel(const __bf16* __restrict__ wht,     // [3H][H] bf16 (K contig)
                const float*  __restrict__ xproj,   // [S][B][3H]
                __bf16*       __restrict__ hb0,     // [B][H] bf16 ping
                __bf16*       __restrict__ hb1,     // [B][H] bf16 pong
                float*        __restrict__ h_all,   // [B][S][H]
                float*        __restrict__ h_final, // [B][H]
                unsigned*     __restrict__ bar_count,
                unsigned*     __restrict__ bar_gen)
{
    extern __shared__ __bf16 lds[];                  // 2 panels of 48 x LDSK
    const int lane = threadIdx.x & 31;
    const int wv   = threadIdx.x >> 5;               // 0..1
    const int j    = blockIdx.x * 2 + wv;            // column tile 0..63

    // ---- Stage both waves' W_h panels into LDS (once for all 512 steps).
    // LDS row (w2*48 + g*16 + c) <- global W_h^T row (g*H + (bx*2+w2)*16 + c).
    for (int row = threadIdx.x; row < 2 * PANEL_ROWS; row += 64) {
        int w2 = row / PANEL_ROWS;
        int r  = row % PANEL_ROWS;
        int g  = r >> 4, c = r & 15;
        int jj = blockIdx.x * 2 + w2;
        const uint4* src = (const uint4*)(wht + ((size_t)(g * GRU_H + jj * 16 + c)) * GRU_H);
        uint4* dst = (uint4*)(lds + (size_t)row * LDSK);
        #pragma unroll 4
        for (int q = 0; q < GRU_H / 8; ++q) dst[q] = src[q];   // 128 x 16B
    }
    __syncthreads();
    const __bf16* panel = lds + (size_t)wv * PANEL_ROWS * LDSK;

    // fp32 recurrent state for this wave's (all-batch x 16-col) slice.
    float hprev[4][8];
    #pragma unroll
    for (int mt = 0; mt < 4; ++mt)
        #pragma unroll
        for (int v = 0; v < 8; ++v) hprev[mt][v] = 0.0f;

    const int col = j * 16 + (lane & 15);

    for (int t = 0; t < GRU_S; ++t) {
        const __bf16* hb_prev = (t & 1) ? hb1 : hb0;
        __bf16*       hb_next = (t & 1) ? hb0 : hb1;

        // Prefetch this step's x_proj rows while the GEMM runs.
        __builtin_prefetch(xproj + ((size_t)t * GRU_B) * GRU_3H + col, 0, 0);

        v8f accZ[4] = {}, accR[4] = {}, accN[4] = {};
        for (int k0 = 0; k0 < GRU_H; k0 += 32) {
            v16bf bz = lds_tile16(panel,  0, k0, lane);
            v16bf br = lds_tile16(panel, 16, k0, lane);
            v16bf bn = lds_tile16(panel, 32, k0, lane);
            #pragma unroll
            for (int mt = 0; mt < 4; ++mt) {
                v16bf a = load_tile16(hb_prev, GRU_H, mt * 16, k0, lane);
                accZ[mt] = wmma_bf16(a, bz, accZ[mt]);
                accR[mt] = wmma_bf16(a, br, accR[mt]);
                accN[mt] = wmma_bf16(a, bn, accN[mt]);
            }
        }

        #pragma unroll
        for (int mt = 0; mt < 4; ++mt) {
            Acc8 cz, cr, cn;
            cz.v = accZ[mt]; cr.v = accR[mt]; cn.v = accN[mt];
            #pragma unroll
            for (int v = 0; v < 8; ++v) {
                int b = mt * 16 + v + ((lane >> 4) << 3);      // batch 0..63
                const float* xp = xproj + ((size_t)t * GRU_B + b) * GRU_3H;
                float xz = xp[col];
                float xr = xp[GRU_H + col];
                float xn = xp[2 * GRU_H + col];
                float z  = 1.0f / (1.0f + __expf(-(xz + cz.f[v])));
                float r  = 1.0f / (1.0f + __expf(-(xr + cr.f[v])));
                float e  = __expf(2.0f * (xn + r * cn.f[v]));
                float nn = 1.0f - 2.0f / (e + 1.0f);           // tanh, safe
                float hnew = (1.0f - z) * nn + z * hprev[mt][v];
                hprev[mt][v] = hnew;
                h_all[((size_t)b * GRU_S + t) * GRU_H + col] = hnew;
                hb_next[(size_t)b * GRU_H + col] = (__bf16)hnew;
                if (t == GRU_S - 1)
                    h_final[(size_t)b * GRU_H + col] = hnew;
            }
        }

        grid_barrier(bar_count, bar_gen, gridDim.x, (unsigned)t);
    }
}

// ---------------------------------------------------------------------------
// Host side
// ---------------------------------------------------------------------------
extern "C" void kernel_launch(void* const* d_in, const int* in_sizes, int n_in,
                              void* d_out, int out_size, void* d_ws, size_t ws_size,
                              hipStream_t stream) {
    (void)in_sizes; (void)n_in; (void)out_size; (void)ws_size;

    const float* x    = (const float*)d_in[0];   // [64][512][512]
    const float* W_in = (const float*)d_in[1];   // [512][3072]
    const float* b_in = (const float*)d_in[2];   // [3072]
    const float* W_h  = (const float*)d_in[3];   // [1024][3072]

    float* h_final = (float*)d_out;                     // [64][1024]
    float* h_all   = (float*)d_out + GRU_B * GRU_H;     // [64][512][1024]

    // Workspace layout (all offsets 256B aligned).
    char* ws = (char*)d_ws;
    __bf16*   wint  = (__bf16*)(ws + 0);           //  3,145,728 B [3072][512]
    __bf16*   wht   = (__bf16*)(ws + 3145728);     //  6,291,456 B [3072][1024]
    __bf16*   xb    = (__bf16*)(ws + 9437184);     // 33,554,432 B [32768][512]
    __bf16*   hb0   = (__bf16*)(ws + 42991616);    //    131,072 B [64][1024]
    __bf16*   hb1   = (__bf16*)(ws + 43122688);    //    131,072 B
    float*    xproj = (float*) (ws + 43253760);    // 402,653,184 B [512][64][3072]
    unsigned* bar   = (unsigned*)(ws + 445906944); //        256 B barrier state

    // Phase 1: bf16 conversion / weight transposition.
    {
        size_t nWin = (size_t)GRU_D * GRU_3H;
        transpose_bf16_kernel<<<(unsigned)((nWin + 255) / 256), 256, 0, stream>>>(
            W_in, wint, GRU_D, GRU_3H);
        size_t nWh = (size_t)GRU_H * GRU_3H;
        transpose_bf16_kernel<<<(unsigned)((nWh + 255) / 256), 256, 0, stream>>>(
            W_h, wht, GRU_H, GRU_3H);
        size_t nX = (size_t)GRU_B * GRU_S * GRU_D;
        convert_bf16_kernel<<<(unsigned)((nX + 255) / 256), 256, 0, stream>>>(
            x, xb, nX);
    }
    // Zero h(-1) and barrier state every call (deterministic across replays).
    hipMemsetAsync(hb0, 0, (size_t)GRU_B * GRU_H * sizeof(__bf16), stream);
    hipMemsetAsync(bar, 0, 256, stream);

    // Phase 2: x_proj GEMM, 103 GFLOP at full parallelism.
    {
        dim3 grid((GRU_B * GRU_S) / 64, GRU_3H / 64); // (512, 48)
        xproj_gemm_kernel<<<grid, 128, 0, stream>>>(xb, wint, b_in, xproj);
    }

    // Phase 3: one persistent kernel runs the whole 512-step scan.
    gru_scan_kernel<<<SCAN_BLOCKS, 64, SCAN_LDS_BYTES, stream>>>(
        wht, xproj, hb0, hb1, h_all, h_final, bar, bar + 16);
}
